// DeconvLayer_50319836840429
// MI455X (gfx1250) — matrix operand order repaired and checked by reading, not compiled, not run on hardware
//
#include <hip/hip_runtime.h>
#include <hip/hip_bf16.h>
#include <math.h>

typedef _Float16 half_t;
typedef __attribute__((ext_vector_type(2)))  _Float16 v2h;
typedef __attribute__((ext_vector_type(16))) _Float16 v16h;
typedef __attribute__((ext_vector_type(8)))  float    v8f;

#define B_    4
#define CIN   256
#define COUT  128
#define H_    64
#define W_    64
#define HW    4096
#define KTOT  2304           // CIN * 9
#define KSTEP 32
#define NKB   72             // KTOT / KSTEP
#define NB_   8              // COUT / 16
#define MT    64             // M positions per workgroup (DCN kernel)
#define MSUB  4              // MT / 16 sub-tiles per wave
#define MTO   128            // M positions per workgroup (offset kernel)
#define ASTRIDE 40           // padded LDS row stride (halves): 80B, 16B-aligned
#define HOUT  128
#define WOUT  128

__device__ __forceinline__ float fast_sigmoid(float v) {
    // v_rcp_f32-based sigmoid (precision ample for sampling masks)
    return __builtin_amdgcn_rcpf(1.0f + __expf(-v));
}

// ---------------------------------------------------------------------------
// Kernel 0a: pack w_dcn (Cout,Cin,3,3) f32 -> fragment-ordered f16
// Bpack[kb][nb][lane][16] ; lane<16 holds K rows 0..15 of the 32-step,
// lane>=16 holds K rows 16..31 (contiguous k order, same as the A-tile LDS reads).
// K index convention: K = ktap*256 + c.
// ---------------------------------------------------------------------------
__global__ __launch_bounds__(256) void pack_w_kernel(
    const float* __restrict__ w_dcn, half_t* __restrict__ Bpack)
{
    int i = blockIdx.x * 256 + threadIdx.x;     // over NKB*NB_*32*16
    if (i >= NKB * NB_ * 32 * 16) return;
    int j    = i & 15;
    int lane = (i >> 4) & 31;
    int nb   = (i >> 9) & 7;
    int kb   = i >> 12;
    int hi   = lane >> 4;
    int o    = nb * 16 + (lane & 15);
    int krow = hi * 16 + j;
    int K    = kb * KSTEP + krow;
    int ktap = K >> 8;        // K / 256
    int c    = K & 255;
    Bpack[i] = (half_t)w_dcn[(o * CIN + c) * 9 + ktap];
}

// ---------------------------------------------------------------------------
// Kernel 0b: pack w_off (27,Cin,3,3) f32 -> fragment-ordered f16, N padded
// to 32 (channels 27..31 zero). Wpack[kb][nb(2)][lane][16].
// ---------------------------------------------------------------------------
__global__ __launch_bounds__(256) void pack_woff_kernel(
    const float* __restrict__ w_off, half_t* __restrict__ Wpack)
{
    int i = blockIdx.x * 256 + threadIdx.x;     // over NKB*2*32*16 = 73728
    if (i >= NKB * 2 * 32 * 16) return;
    int j    = i & 15;
    int lane = (i >> 4) & 31;
    int nb   = (i >> 9) & 1;
    int kb   = i >> 10;
    int hi   = lane >> 4;
    int o    = nb * 16 + (lane & 15);
    int krow = hi * 16 + j;
    int K    = kb * KSTEP + krow;
    int ktap = K >> 8;
    int c    = K & 255;
    Wpack[i] = (o < 27) ? (half_t)w_off[(o * CIN + c) * 9 + ktap] : (half_t)0.0f;
}

// ---------------------------------------------------------------------------
// Kernel 1: offset conv as WMMA GEMM (M=B*HW, K=2304, N=27 padded to 32).
// Workgroup = 8 waves, M-tile = 128 positions; wave -> (nb = N-tile 0..1,
// 2 M sub-tiles). A tile = shifted-window copy of x (im2col) with clamped
// unconditional loads. Epilogue adds b_off, routes to dy/dx/sigmoid(mask).
// ---------------------------------------------------------------------------
__global__ __launch_bounds__(256) void offset_wmma_kernel(
    const float* __restrict__ x, const half_t* __restrict__ Wpack,
    const float* __restrict__ b_off,
    float* __restrict__ dy, float* __restrict__ dx, float* __restrict__ mk)
{
    __shared__ alignas(16) half_t A_lds[MTO][ASTRIDE];

    int tile = blockIdx.x;              // 0 .. B*HW/MTO - 1
    int b    = tile / (HW / MTO);
    int p0   = (tile - b * (HW / MTO)) * MTO;
    int tid  = threadIdx.x;
    int lane = tid & 31;
    int wv   = tid >> 5;
    int nb   = wv & 1;                  // N tile (0: ch 0..15, 1: ch 16..31)
    int mgb  = (wv >> 1) * 2;           // first of 2 M sub-tiles for this wave
    int hi   = lane >> 4;
    int M    = lane & 15;

    v8f acc0 = {}, acc1 = {};
    const float* xb = x + (size_t)b * CIN * HW;
    const half_t* bp = Wpack + ((size_t)nb * 32 + lane) * 16;

    for (int ktap = 0; ktap < 9; ++ktap) {
        int ty = ktap / 3 - 1, tx = ktap % 3 - 1;   // loop-invariant
        for (int cb = 0; cb < 8; ++cb) {
            int cbase = cb * KSTEP;

            // A-tile: im2col window copy; clamped unconditional loads
            #pragma unroll
            for (int i = 0; i < MTO * 16 / 256; ++i) {
                int pr = tid + i * 256;
                int m  = pr >> 4;
                int cc = (pr & 15) * 2;
                int p  = p0 + m;
                int hh = p >> 6, ww = p & 63;
                int yy = hh + ty, xx = ww + tx;
                bool ok = ((unsigned)yy < (unsigned)H_) &&
                          ((unsigned)xx < (unsigned)W_);
                int yc = min(max(yy, 0), H_ - 1);
                int xc = min(max(xx, 0), W_ - 1);
                const float* src = xb + (size_t)(cbase + cc) * HW + yc * W_ + xc;
                float la = src[0], lb = src[HW];
                float va = ok ? la : 0.0f;
                float vb = ok ? lb : 0.0f;
                v2h pk = { (half_t)va, (half_t)vb };
                *(v2h*)&A_lds[m][cc] = pk;
            }
            __syncthreads();

            v16h bf = *(const v16h*)bp;
            __builtin_prefetch((const void*)(bp + (size_t)2 * 32 * 16), 0, 3);

            union AFrag { v16h h; float4 q[2]; } af;
            #pragma unroll
            for (int t = 0; t < 2; ++t) {
                int row = (mgb + t) * 16 + M;
                af.q[0] = *(const float4*)&A_lds[row][16 * hi];
                af.q[1] = *(const float4*)&A_lds[row][16 * hi + 8];
                v8f* accp = t ? &acc1 : &acc0;
                *accp = __builtin_amdgcn_wmma_f32_16x16x32_f16(
                            false, af.h, false, bf, (short)0, *accp, false, false);
            }
            __syncthreads();
            bp += (size_t)2 * 32 * 16;    // next K-step fragment block
        }
    }

    // epilogue: route channel o to dy / dx / mask (resolve destination once)
    int o = nb * 16 + M;
    if (o < 27) {
        float bo = b_off[o];
        bool  is_mask = (o >= 18);
        int   k   = is_mask ? (o - 18) : (o >> 1);
        float* d  = is_mask ? mk : ((o & 1) ? dx : dy);
        float* dst = d + ((size_t)b * 9 + k) * HW;
        #pragma unroll
        for (int t = 0; t < 2; ++t) {
            const v8f* accp = t ? &acc1 : &acc0;
            #pragma unroll
            for (int r = 0; r < 8; ++r) {
                int p = p0 + (mgb + t) * 16 + r + 8 * hi;
                float v = (*accp)[r] + bo;
                dst[p] = is_mask ? fast_sigmoid(v) : v;
            }
        }
    }
}

// ---------------------------------------------------------------------------
// Kernel 2: fused deformable gather + GEMM via v_wmma_f32_16x16x32_f16.
// Workgroup = 8 waves; M-tile = 64 positions. Wave w owns N-tile w and
// 4 accumulators (one per 16-row M sub-tile): 4 WMMAs per K-step reusing
// a single B fragment; A fragments are 2x ds_load_b128 each.
// ---------------------------------------------------------------------------
__global__ __launch_bounds__(256) void dcn_wmma_kernel(
    const float* __restrict__ x, const half_t* __restrict__ Bpack,
    const float* __restrict__ dy, const float* __restrict__ dx,
    const float* __restrict__ mk,
    const float* __restrict__ b_dcn, const float* __restrict__ gamma,
    const float* __restrict__ beta, const float* __restrict__ rmean,
    const float* __restrict__ rvar, float* __restrict__ pre)
{
    __shared__ float  sw[MT][9][4];     // bilinear weights * mask * valid
    __shared__ int    sidx[MT][9][4];   // clamped in-image linear offsets
    __shared__ alignas(16) half_t A_lds[MT][ASTRIDE];

    int tile = blockIdx.x;              // 0 .. B*HW/MT - 1
    int b    = tile / (HW / MT);
    int p0   = (tile - b * (HW / MT)) * MT;
    int tid  = threadIdx.x;
    int lane = tid & 31;
    int wv   = tid >> 5;                // wave id = N tile
    int hi   = lane >> 4;
    int M    = lane & 15;

    // --- precompute sampling coefficients: MT positions x 9 taps ---
    for (int t = tid; t < MT * 9; t += 256) {
        int m = t / 9, k = t - m * 9;
        int p  = p0 + m;
        int hh = p >> 6, ww = p & 63;
        size_t base = ((size_t)b * 9 + k) * HW + p;
        float py = dy[base] + (float)hh + (float)(k / 3 - 1);
        float px = dx[base] + (float)ww + (float)(k % 3 - 1);
        float msk = mk[base];
        float fy = floorf(py), fx = floorf(px);
        int y0 = (int)fy, x0 = (int)fx;
        float gy = py - fy, gx = px - fx;
        float wy[2] = {1.0f - gy, gy};
        float wx[2] = {1.0f - gx, gx};
        #pragma unroll
        for (int q = 0; q < 4; ++q) {
            int yi = y0 + (q >> 1);
            int xi = x0 + (q & 1);
            bool valid = (yi >= 0) && (yi < H_) && (xi >= 0) && (xi < W_);
            int yc = min(max(yi, 0), H_ - 1);
            int xc = min(max(xi, 0), W_ - 1);
            sidx[m][k][q] = yc * W_ + xc;
            sw[m][k][q]   = valid ? wy[q >> 1] * wx[q & 1] * msk : 0.0f;
        }
    }
    __syncthreads();

    v8f acc0 = {}, acc1 = {}, acc2 = {}, acc3 = {};
    const float* xb = x + (size_t)b * CIN * HW;
    const half_t* bp = Bpack + ((size_t)wv * 32 + lane) * 16;

    for (int ktap = 0; ktap < 9; ++ktap) {
        for (int cb = 0; cb < 8; ++cb) {
            int cbase = cb * KSTEP;

            // cooperative A-tile gather: MT pos x 32 ch; channel-pair/thread
            #pragma unroll
            for (int i = 0; i < 4; ++i) {
                int pr = tid + i * 256;
                int m  = pr >> 4;
                int cc = (pr & 15) * 2;
                const float* xc0 = xb + (size_t)(cbase + cc) * HW;
                const float* xc1 = xc0 + HW;
                float w0 = sw[m][ktap][0], w1 = sw[m][ktap][1];
                float w2 = sw[m][ktap][2], w3 = sw[m][ktap][3];
                int i0 = sidx[m][ktap][0], i1 = sidx[m][ktap][1];
                int i2 = sidx[m][ktap][2], i3 = sidx[m][ktap][3];
                float va = w0*xc0[i0] + w1*xc0[i1] + w2*xc0[i2] + w3*xc0[i3];
                float vb = w0*xc1[i0] + w1*xc1[i1] + w2*xc1[i2] + w3*xc1[i3];
                v2h pk = { (half_t)va, (half_t)vb };
                *(v2h*)&A_lds[m][cc] = pk;
            }
            __syncthreads();

            // B fragment: contiguous 32B per lane (fragment-ordered pack)
            v16h bf = *(const v16h*)bp;
            __builtin_prefetch((const void*)(bp + (size_t)NB_ * 32 * 16), 0, 3);

            // A fragments: two aligned 16B LDS loads per sub-tile
            union AFrag { v16h h; float4 q[2]; } af;
            #pragma unroll
            for (int t = 0; t < MSUB; ++t) {
                int row = t * 16 + M;
                af.q[0] = *(const float4*)&A_lds[row][16 * hi];
                af.q[1] = *(const float4*)&A_lds[row][16 * hi + 8];
                v8f* accp = (t == 0) ? &acc0 : (t == 1) ? &acc1
                          : (t == 2) ? &acc2 : &acc3;
                *accp = __builtin_amdgcn_wmma_f32_16x16x32_f16(
                            false, af.h, false, bf, (short)0, *accp, false, false);
            }
            __syncthreads();   // protect A_lds before next fill
            bp += (size_t)NB_ * 32 * 16;
        }
    }

    // --- epilogue: +b_dcn, BN (inference), ReLU; C/D lane map from ISA ---
    int o = wv * 16 + (lane & 15);
    float rstd = rsqrtf(rvar[o] + 1e-5f);
    float sc = gamma[o] * rstd;
    float sh = beta[o] - gamma[o] * rmean[o] * rstd + b_dcn[o] * sc;
    float* outp = pre + ((size_t)b * COUT + o) * HW;
    #pragma unroll
    for (int t = 0; t < MSUB; ++t) {
        const v8f* accp = (t == 0) ? &acc0 : (t == 1) ? &acc1
                        : (t == 2) ? &acc2 : &acc3;
        #pragma unroll
        for (int r = 0; r < 8; ++r) {
            int p = p0 + t * 16 + r + 8 * hi;   // M = r + 8*(lane>=16)
            float v = (*accp)[r] * sc + sh;
            outp[p] = v > 0.0f ? v : 0.0f;
        }
    }
}

// ---------------------------------------------------------------------------
// Kernel 3: x2 align-corners bilinear upsample (64 -> 128).
// ---------------------------------------------------------------------------
__global__ __launch_bounds__(256) void upsample_kernel(
    const float* __restrict__ pre, float* __restrict__ out)
{
    int i = blockIdx.x * 256 + threadIdx.x;
    if (i >= B_ * COUT * HOUT * WOUT) return;
    int xo = i & 127;
    int yo = (i >> 7) & 127;
    int oc = (i >> 14) & 127;
    int b  = i >> 21;
    const float s = 63.0f / 127.0f;
    float py = yo * s, px = xo * s;
    float fy = floorf(py), fx = floorf(px);
    int y0 = (int)fy, x0 = (int)fx;
    int y1 = min(y0 + 1, H_ - 1), x1 = min(x0 + 1, W_ - 1);
    float gy = py - fy, gx = px - fx;
    const float* src = pre + ((size_t)b * COUT + oc) * HW;
    float v = src[y0 * W_ + x0] * (1.0f - gy) * (1.0f - gx)
            + src[y0 * W_ + x1] * (1.0f - gy) * gx
            + src[y1 * W_ + x0] * gy * (1.0f - gx)
            + src[y1 * W_ + x1] * gy * gx;
    out[i] = v;
}

// ---------------------------------------------------------------------------
extern "C" void kernel_launch(void* const* d_in, const int* in_sizes, int n_in,
                              void* d_out, int out_size, void* d_ws, size_t ws_size,
                              hipStream_t stream) {
    const float* x     = (const float*)d_in[0];
    const float* w_off = (const float*)d_in[1];
    const float* b_off = (const float*)d_in[2];
    const float* w_dcn = (const float*)d_in[3];
    const float* b_dcn = (const float*)d_in[4];
    const float* gamma = (const float*)d_in[5];
    const float* beta  = (const float*)d_in[6];
    const float* rmean = (const float*)d_in[7];
    const float* rvar  = (const float*)d_in[8];
    float* out = (float*)d_out;

    // workspace layout (each section 256B-aligned)
    const size_t BPACK_BYTES = (size_t)NKB * NB_ * 32 * 16 * sizeof(half_t); // 589824
    const size_t WPACK_BYTES = (size_t)NKB * 2 * 32 * 16 * sizeof(half_t);   // 147456
    const size_t OFF_BYTES   = (size_t)B_ * 9 * HW * sizeof(float);          // 589824
    char* ws = (char*)d_ws;
    half_t* Bpack = (half_t*)ws;
    half_t* Wpack = (half_t*)(ws + BPACK_BYTES);
    float*  dy    = (float*)(ws + BPACK_BYTES + WPACK_BYTES);
    float*  dx    = (float*)(ws + BPACK_BYTES + WPACK_BYTES + OFF_BYTES);
    float*  mk    = (float*)(ws + BPACK_BYTES + WPACK_BYTES + 2 * OFF_BYTES);
    float*  pre   = (float*)(ws + BPACK_BYTES + WPACK_BYTES + 3 * OFF_BYTES);

    pack_w_kernel<<<(NKB * NB_ * 32 * 16 + 255) / 256, 256, 0, stream>>>(w_dcn, Bpack);
    pack_woff_kernel<<<(NKB * 2 * 32 * 16 + 255) / 256, 256, 0, stream>>>(w_off, Wpack);
    offset_wmma_kernel<<<B_ * HW / MTO, 256, 0, stream>>>(x, Wpack, b_off, dy, dx, mk);
    dcn_wmma_kernel<<<B_ * HW / MT, 256, 0, stream>>>(
        x, Bpack, dy, dx, mk, b_dcn, gamma, beta, rmean, rvar, pre);
    upsample_kernel<<<(B_ * COUT * HOUT * WOUT + 255) / 256, 256, 0, stream>>>(pre, out);
}
